// BDHModel_19215683682991
// MI455X (gfx1250) — compile-verified, matrix operand order checked by reading, not run on hardware
//
#include <hip/hip_runtime.h>
#include <hip/hip_bf16.h>
#include <math.h>

#define NDIM   2048
#define KTOP   20
#define TLEN   128
#define LRCONS 0.01f
#define EPSC   1e-8f
#define MAXIDX 64

typedef __attribute__((ext_vector_type(16))) _Float16 v16h;
typedef __attribute__((ext_vector_type(8)))  float    v8f;

// ---------- helpers ----------
__device__ __forceinline__ uint32_t f32key(float f) {
  uint32_t b = __float_as_uint(f);
  return (b & 0x80000000u) ? ~b : (b | 0x80000000u);  // monotone order-preserving map
}

__device__ __forceinline__ float blockReduceSum(float v, float* red) {
  __syncthreads();                         // protect red[] reuse across calls
  #pragma unroll
  for (int off = 16; off > 0; off >>= 1)
    v += __shfl_down(v, off, 32);          // wave32 reduce
  int lane = threadIdx.x & 31, wave = threadIdx.x >> 5;
  if (lane == 0) red[wave] = v;
  __syncthreads();
  if (wave == 0) {
    float r = (lane < ((int)blockDim.x >> 5)) ? red[lane] : 0.f;
    #pragma unroll
    for (int off = 16; off > 0; off >>= 1)
      r += __shfl_down(r, off, 32);
    if (lane == 0) red[0] = r;
  }
  __syncthreads();
  return red[0];
}

// ---------- 1) gather + exact top-K threshold -> binary acts ----------
__global__ void __launch_bounds__(256)
acts_kernel(const int* __restrict__ tokens, const float* __restrict__ proj,
            _Float16* __restrict__ actsDense, int* __restrict__ idxList,
            int* __restrict__ idxCnt)
{
  __shared__ int s_cnt;
  __shared__ int s_num;
  const int t   = blockIdx.x;
  const int tid = threadIdx.x;
  const long row = (long)tokens[t] * NDIM;

  uint32_t keys[8];
  #pragma unroll
  for (int k = 0; k < 8; ++k)
    keys[k] = f32key(proj[row + tid + k * 256]);

  // exact K-th largest via monotone binary search on 32-bit key space
  unsigned long long lo = 0ull, hi = 0x100000000ull;
  while (hi - lo > 1ull) {
    unsigned long long mid = lo + ((hi - lo) >> 1);
    if (tid == 0) s_cnt = 0;
    __syncthreads();
    int c = 0;
    #pragma unroll
    for (int k = 0; k < 8; ++k)
      c += ((unsigned long long)keys[k] >= mid) ? 1 : 0;
    atomicAdd(&s_cnt, c);
    __syncthreads();
    int tot = s_cnt;
    __syncthreads();
    if (tot >= KTOP) lo = mid; else hi = mid;
  }
  const uint32_t thr = (uint32_t)lo;

  if (tid == 0) s_num = 0;
  __syncthreads();
  #pragma unroll
  for (int k = 0; k < 8; ++k) {
    int j = tid + k * 256;
    bool act = keys[k] >= thr;               // matches (raw >= thr), ties included
    actsDense[(long)t * NDIM + j] = act ? (_Float16)1.0f : (_Float16)0.0f;
    if (act) {
      int p = atomicAdd(&s_num, 1);
      if (p < MAXIDX) idxList[t * MAXIDX + p] = j;
    }
  }
  __syncthreads();
  if (tid == 0) idxCnt[t] = (s_num < MAXIDX) ? s_num : MAXIDX;
}

// ---------- 2) build transposed sigma copy (inputs are read-only) ----------
__global__ void __launch_bounds__(256)
transpose_kernel(const float* __restrict__ src, float* __restrict__ dst)
{
  __shared__ float tile[32][33];
  const int bx = blockIdx.x * 32, by = blockIdx.y * 32;
  const int tx = threadIdx.x, ty = threadIdx.y;   // 32 x 8
  #pragma unroll
  for (int yy = 0; yy < 32; yy += 8)
    tile[ty + yy][tx] = src[(long)(by + ty + yy) * NDIM + bx + tx];
  __syncthreads();
  #pragma unroll
  for (int yy = 0; yy < 32; yy += 8)
    dst[(long)(bx + ty + yy) * NDIM + by + tx] = tile[tx][ty + yy];
}

// ---------- 3) plasticity==1: persistent single-WGP sequential scan ----------
// pred = sum of the ~20 active columns of sigma (read coalesced via sigmaT);
// update touches only cn*cc (~400) entries per step.
__global__ void __launch_bounds__(1024)
scan_kernel(const int* __restrict__ plast, float* __restrict__ sigmaT,
            const _Float16* __restrict__ actsDense,
            const int* __restrict__ idxList, const int* __restrict__ idxCnt,
            float* __restrict__ out)
{
  if (*plast == 0) return;                 // wave-uniform gate
  __shared__ float red[32];
  __shared__ int s_ic[MAXIDX];
  __shared__ int s_in[MAXIDX];
  __shared__ int s_cc, s_cn;
  const int tid = threadIdx.x;
  const float sqrtk = 4.47213595499958f;   // sqrt(K=20)

  for (int t = 0; t < TLEN - 1; ++t) {
    if (tid == 0) { s_cc = idxCnt[t]; s_cn = idxCnt[t + 1]; }
    __syncthreads();
    const int cc = s_cc, cn = s_cn;
    if (tid < cc) s_ic[tid] = idxList[t * MAXIDX + tid];
    if (tid < cn) s_in[tid] = idxList[(t + 1) * MAXIDX + tid];
    __syncthreads();

    // each thread owns rows tid and tid+1024 of pred
    float p0 = 0.f, p1 = 0.f;
    for (int k = 0; k < cc; ++k) {
      const float* col = sigmaT + (long)s_ic[k] * NDIM;
      p0 += col[tid];
      p1 += col[tid + 1024];
    }
    const float xn0 = (float)actsDense[(long)(t + 1) * NDIM + tid];
    const float xn1 = (float)actsDense[(long)(t + 1) * NDIM + tid + 1024];

    const float S2 = blockReduceSum(p0 * p0 + p1 * p1, red);
    const float SD = blockReduceSum(p0 * xn0 + p1 * xn1, red);

    if (tid == 0) {
      float pn = sqrtf(S2);
      float ov = SD / (pn * sqrtk + EPSC);
      out[t] = (pn > 0.f) ? (1.f - ov) : 1.f;
    }
    __syncthreads();

    // sigma update: only entries where outer(x_next, x_curr) != 0 can change
    const int tot = cn * cc;
    for (int p = tid; p < tot; p += 1024) {
      int a = s_in[p / cc];                // row in sigma  (x_next index)
      int b = s_ic[p % cc];                // col in sigma  (x_curr index)
      float* sp = sigmaT + (long)b * NDIM + a;  // transposed storage
      float v = *sp + LRCONS;
      *sp = (v > 1.f) ? 1.f : ((v < 0.f) ? 0.f : v);
    }
    __threadfence();
    __syncthreads();
  }
}

// ---------- 4) plasticity==0: dense preds = x_curr @ sigma^T via WMMA ----------
// One wave computes a 16x16 tile of preds[t][i]; acts are exact in f16.
__global__ void __launch_bounds__(32)
fb_gemm(const int* __restrict__ plast, const float* __restrict__ sigma,
        const _Float16* __restrict__ actsDense,
        float* __restrict__ sq, float* __restrict__ dt)
{
  if (*plast != 0) return;                 // wave-uniform gate, EXEC stays all-1s for WMMA
  const int t0 = blockIdx.x * 16;
  const int i0 = blockIdx.y * 16;
  const int lane = threadIdx.x;            // 0..31 (wave32)
  const int m  = lane & 15;
  const int hb = lane >> 4;

  v8f c = {};
  for (int j0 = 0; j0 < NDIM; j0 += 32) {
    v16h a, b;
    // ISA 16-bit A 16x32 layout: lane<16 -> M=lane, K in {0..7,16..23};
    // lane>=16 -> M=lane-16, K in {8..15,24..31}. B (32x16) mirrored with N=m.
    #pragma unroll
    for (int e = 0; e < 16; ++e) {
      int k = (e < 8) ? (e + hb * 8) : (8 + e + hb * 8);   // e>=8: 16+(e-8)+hb*8
      a[e] = actsDense[(long)(t0 + m) * NDIM + j0 + k];
      b[e] = (_Float16)sigma[(long)(i0 + m) * NDIM + j0 + k];  // B[k][n] = sigma[i0+n][j0+k]
    }
    c = __builtin_amdgcn_wmma_f32_16x16x32_f16(
        /*neg_a=*/false, a, /*neg_b=*/false, b,
        /*c_mod=*/(short)0, c, /*reuse_a=*/false, /*reuse_b=*/false);
  }

  // C/D layout: VGPR v -> M = v + 8*hb, N = m
  #pragma unroll
  for (int v = 0; v < 8; ++v) {
    int t = t0 + v + 8 * hb;
    int i = i0 + m;
    if (t < TLEN - 1) {
      float p = c[v];
      atomicAdd(&sq[t], p * p);
      atomicAdd(&dt[t], p * (float)actsDense[(long)(t + 1) * NDIM + i]);
    }
  }
}

__global__ void fb_final(const int* __restrict__ plast, const float* __restrict__ sq,
                         const float* __restrict__ dt, const int* __restrict__ idxCnt,
                         float* __restrict__ out)
{
  if (*plast != 0) return;
  int t = blockIdx.x * blockDim.x + threadIdx.x;
  if (t < TLEN - 1) {
    float nn = sqrtf((float)idxCnt[t + 1]);           // ||x_next||
    float ov = dt[t] / (sqrtf(sq[t]) * nn + EPSC);
    out[t] = 1.f - ov;
  }
}

// ---------- launch ----------
extern "C" void kernel_launch(void* const* d_in, const int* in_sizes, int n_in,
                              void* d_out, int out_size, void* d_ws, size_t ws_size,
                              hipStream_t stream)
{
  const int*   tokens = (const int*)d_in[0];
  const float* proj   = (const float*)d_in[1];
  const float* sigma  = (const float*)d_in[2];
  const int*   plast  = (const int*)d_in[3];
  float* out = (float*)d_out;

  char* ws = (char*)d_ws;
  float*    sigmaT    = (float*)ws;     ws += (size_t)NDIM * NDIM * sizeof(float);
  _Float16* actsDense = (_Float16*)ws;  ws += (size_t)TLEN * NDIM * sizeof(_Float16);
  int*      idxList   = (int*)ws;       ws += (size_t)TLEN * MAXIDX * sizeof(int);
  int*      idxCnt    = (int*)ws;       ws += (size_t)TLEN * sizeof(int);
  float*    sq        = (float*)ws;     ws += (size_t)TLEN * sizeof(float);
  float*    dt        = (float*)ws;     ws += (size_t)TLEN * sizeof(float);

  // rebuild all scratch every call -> deterministic under graph replay
  hipMemsetAsync(sq, 0, 2 * TLEN * sizeof(float), stream);       // sq and dt contiguous
  transpose_kernel<<<dim3(NDIM / 32, NDIM / 32), dim3(32, 8), 0, stream>>>(sigma, sigmaT);
  acts_kernel<<<TLEN, 256, 0, stream>>>(tokens, proj, actsDense, idxList, idxCnt);

  // plasticity == 1 path (self-gated on device scalar)
  scan_kernel<<<1, 1024, 0, stream>>>(plast, sigmaT, actsDense, idxList, idxCnt, out);

  // plasticity == 0 path (self-gated; WMMA dense GEMM)
  fb_gemm<<<dim3(TLEN / 16, NDIM / 16), 32, 0, stream>>>(plast, sigma, actsDense, sq, dt);
  fb_final<<<1, 128, 0, stream>>>(plast, sq, dt, idxCnt, out);
}
